// WMG_Network_48524540510706
// MI455X (gfx1250) — compile-verified, matrix operand order checked by reading, not run on hardware
//
#include <hip/hip_runtime.h>
#include <hip/hip_bf16.h>

// ---------------------------------------------------------------------------
// WMG transformer forward for gfx1250 (MI455X).
// All GEMMs + attention use v_wmma_f32_16x16x32_bf16 (wave32, 16x16x32).
// GEMM waves are register-blocked 16x64 (A reused across 4 WMMAs).
// ---------------------------------------------------------------------------

#define S_LEN 2048
#define OBS_N 256
#define SV_N  2047
#define D_N   256
#define H_N   8
#define HD_N  32
#define L_N   2
#define FF_N  1024
#define ACH_N 512
#define A_N   16

typedef __attribute__((ext_vector_type(16))) __bf16 v16bf;
typedef __attribute__((ext_vector_type(8)))  __bf16 v8bf;
typedef __attribute__((ext_vector_type(8)))  float  v8f;

// Load 16 bf16 in WMMA 16-bit A/B per-lane layout.
// Caller passes p already offset by (half ? 8 : 0); elements 0..7 come from
// p[0..7] and elements 8..15 from p[16..23]  (ISA 7.12.2 16-bit A 16x32 table).
__device__ inline v16bf load16(const __bf16* p) {
  v8bf lo = *(const v8bf*)(p);
  v8bf hi = *(const v8bf*)(p + 16);
  v16bf r;
#pragma unroll
  for (int e = 0; e < 8; ++e) { r[e] = lo[e]; r[e + 8] = hi[e]; }
  return r;
}

__device__ inline v8f wmma_bf16(v16bf a, v16bf b, v8f c) {
  return __builtin_amdgcn_wmma_f32_16x16x32_bf16(
      /*neg_a=*/false, a, /*neg_b=*/false, b,
      /*c_mod=*/(short)0, c, /*reuse_a=*/false, /*reuse_b=*/false);
}

__device__ inline float gelu_t(float x) {
  const float k0 = 0.7978845608028654f, k1 = 0.044715f;
  return 0.5f * x * (1.0f + tanhf(k0 * (x + k1 * x * x * x)));
}

// ---------------------------------------------------------------------------
// Weight cast: fp32 (K x N) row-major  ->  bf16 (N x K) row-major.
// Transposed layout makes the WMMA B operand load as contiguous b128s.
// ---------------------------------------------------------------------------
__global__ __launch_bounds__(256)
void cast_transpose_kernel(const float* __restrict__ W, __bf16* __restrict__ Wt,
                           int K, int N) {
  int total = K * N;
  for (int idx = blockIdx.x * blockDim.x + threadIdx.x; idx < total;
       idx += gridDim.x * blockDim.x) {
    int k = idx / N, n = idx % N;
    Wt[(size_t)n * K + k] = (__bf16)W[idx];
  }
}

// ---------------------------------------------------------------------------
// Embedding: row 0 = obs @ W_core + b_core;
//            row s = old_vectors[s-1] @ W_memo[:256] + W_memo[256+s-1] + b_memo
// ---------------------------------------------------------------------------
__global__ __launch_bounds__(256)
void embed_kernel(const float* __restrict__ obs, const float* __restrict__ oldv,
                  const float* __restrict__ Wc, const float* __restrict__ bc,
                  const float* __restrict__ Wm, const float* __restrict__ bm,
                  float* __restrict__ xf, __bf16* __restrict__ xh) {
  __shared__ float row[D_N];
  int s = blockIdx.x, d = threadIdx.x;
  const float* in = (s == 0) ? obs : (oldv + (size_t)(s - 1) * OBS_N);
  row[d] = in[d];
  __syncthreads();
  const float* W = (s == 0) ? Wc : Wm;
  float acc = 0.f;
#pragma unroll 8
  for (int j = 0; j < OBS_N; ++j) acc += row[j] * W[(size_t)j * D_N + d];
  if (s == 0) acc += bc[d];
  else        acc += Wm[(size_t)(OBS_N + s - 1) * D_N + d] + bm[d];
  xf[(size_t)s * D_N + d] = acc;
  xh[(size_t)s * D_N + d] = (__bf16)acc;
}

// ---------------------------------------------------------------------------
// Register-blocked WMMA GEMM:
//   C(MxN) = act( (A(MxK) @ Bt^T (+bias)) * scale (+resid) )
// Block = 4 waves (128 thr) covering a 64(M) x 64(N) tile.
// Each wave computes one 16x64 strip: A loaded once per K-step, reused across
// 4 back-to-back WMMAs; 10 b128 loads per 4 WMMAs.
// A row-major MxK bf16, Bt is N x K bf16 (pre-transposed).
// act: 0=none, 1=gelu, 2=tanh. Any of Cf / Cb / CbT may be null.
// Requires M % 64 == 0, N % 64 == 0, K % 32 == 0.
// ---------------------------------------------------------------------------
__global__ __launch_bounds__(128)
void gemm_bf16_wmma(const __bf16* __restrict__ A, const __bf16* __restrict__ Bt,
                    const float* __restrict__ bias, const float* __restrict__ resid,
                    float scale, int act,
                    float* __restrict__ Cf, __bf16* __restrict__ Cb,
                    __bf16* __restrict__ CbT,
                    int M, int N, int K) {
  int lane = threadIdx.x & 31;
  int wave = threadIdx.x >> 5;
  int half = lane >> 4;
  int lid  = lane & 15;
  int n0 = blockIdx.x * 64;
  int m0 = blockIdx.y * 64 + wave * 16;
  int off0 = half ? 8 : 0;
  const __bf16* arow = A  + (size_t)(m0 + lid) * K + off0;
  const __bf16* brow = Bt + (size_t)(n0 + lid) * K + off0;
  const size_t bstep = (size_t)16 * K;

  v8f acc0 = {}, acc1 = {}, acc2 = {}, acc3 = {};
  for (int k = 0; k < K; k += 32) {
    // pull next K-slab toward the WGP while this step's WMMAs execute
    __builtin_prefetch(arow + k + 128, 0, 1);
    __builtin_prefetch(brow + k + 128, 0, 1);
    v16bf a  = load16(arow + k);
    v16bf b0 = load16(brow + k);
    v16bf b1 = load16(brow + bstep + k);
    v16bf b2 = load16(brow + 2 * bstep + k);
    v16bf b3 = load16(brow + 3 * bstep + k);
    acc0 = wmma_bf16(a, b0, acc0);
    acc1 = wmma_bf16(a, b1, acc1);
    acc2 = wmma_bf16(a, b2, acc2);
    acc3 = wmma_bf16(a, b3, acc3);
  }

  v8f accs[4] = {acc0, acc1, acc2, acc3};
#pragma unroll
  for (int j = 0; j < 4; ++j) {
#pragma unroll
    for (int i = 0; i < 8; ++i) {
      int m = m0 + i + 8 * half;  // C layout: vgpr i -> row i (lanes 0-15), i+8 (16-31)
      int n = n0 + j * 16 + lid;
      float v = accs[j][i];
      if (bias)  v += bias[n];
      v *= scale;
      if (resid) v += resid[(size_t)m * N + n];
      if (act == 1)      v = gelu_t(v);
      else if (act == 2) v = tanhf(v);
      if (Cf)  Cf[(size_t)m * N + n] = v;
      if (Cb)  Cb[(size_t)m * N + n] = (__bf16)v;
      if (CbT) CbT[(size_t)n * M + m] = (__bf16)v;
    }
  }
}

// ---------------------------------------------------------------------------
// Flash-style attention. One wave per (head, 16-query tile).
// Q,K row-major (S x 256) bf16 (scale pre-folded into Q). Vt is (256 x S)
// bf16 (dim-major) so the P@V B-operand loads contiguously.
// Online softmax: row r of the score tile lives in one 16-lane half-wave,
// reduced via shfl_xor butterflies of width 16.
// ---------------------------------------------------------------------------
__global__ __launch_bounds__(32)
void attn_wmma(const __bf16* __restrict__ Q, const __bf16* __restrict__ Kb,
               const __bf16* __restrict__ Vt, __bf16* __restrict__ Ctx) {
  __shared__ __align__(16) __bf16 pbuf[16][32];
  int lane = threadIdx.x;
  int half = lane >> 4;
  int lid  = lane & 15;
  int h  = blockIdx.x;
  int qt = blockIdx.y;
  int off0 = half ? 8 : 0;
  int hc = h * HD_N;

  v16bf qa = load16(Q + (size_t)(qt * 16 + lid) * D_N + hc + off0);
  v8f acc0 = {}, acc1 = {};
  float mrun[8], lrun[8];
#pragma unroll
  for (int i = 0; i < 8; ++i) { mrun[i] = -1e30f; lrun[i] = 0.f; }

  for (int kc = 0; kc < S_LEN; kc += 32) {
    // scores: S = Q(16x32) x K_chunk^T(32x32) as two 16x16 WMMA tiles
    v16bf b0 = load16(Kb + (size_t)(kc + lid) * D_N + hc + off0);
    v16bf b1 = load16(Kb + (size_t)(kc + 16 + lid) * D_N + hc + off0);
    v8f z0 = {}, z1 = {};
    v8f s0 = wmma_bf16(qa, b0, z0);
    v8f s1 = wmma_bf16(qa, b1, z1);

#pragma unroll
    for (int i = 0; i < 8; ++i) {
      float mx = fmaxf(s0[i], s1[i]);
      mx = fmaxf(mx, __shfl_xor(mx, 1, 16));
      mx = fmaxf(mx, __shfl_xor(mx, 2, 16));
      mx = fmaxf(mx, __shfl_xor(mx, 4, 16));
      mx = fmaxf(mx, __shfl_xor(mx, 8, 16));
      float mnew = fmaxf(mrun[i], mx);
      float corr = __expf(mrun[i] - mnew);
      float p0 = __expf(s0[i] - mnew);
      float p1 = __expf(s1[i] - mnew);
      float sum = p0 + p1;
      sum += __shfl_xor(sum, 1, 16);
      sum += __shfl_xor(sum, 2, 16);
      sum += __shfl_xor(sum, 4, 16);
      sum += __shfl_xor(sum, 8, 16);
      lrun[i] = lrun[i] * corr + sum;
      mrun[i] = mnew;
      acc0[i] *= corr;
      acc1[i] *= corr;
      int r = i + 8 * half;
      pbuf[r][lid]      = (__bf16)p0;
      pbuf[r][lid + 16] = (__bf16)p1;
    }
    __syncthreads();
    // P (16x32) as A operand from LDS; V chunk as two 32x16 B operands
    v16bf pa  = load16(&pbuf[lid][0] + off0);
    v16bf bv0 = load16(Vt + (size_t)(hc + lid) * S_LEN + kc + off0);
    v16bf bv1 = load16(Vt + (size_t)(hc + 16 + lid) * S_LEN + kc + off0);
    acc0 = wmma_bf16(pa, bv0, acc0);
    acc1 = wmma_bf16(pa, bv1, acc1);
    __syncthreads();
  }

#pragma unroll
  for (int i = 0; i < 8; ++i) {
    int m = qt * 16 + i + 8 * half;
    float inv = 1.f / lrun[i];
    Ctx[(size_t)m * D_N + hc + lid]      = (__bf16)(acc0[i] * inv);
    Ctx[(size_t)m * D_N + hc + 16 + lid] = (__bf16)(acc1[i] * inv);
  }
}

// ---------------------------------------------------------------------------
// Row LayerNorm over D=256; writes fp32 residual + bf16 activation copies.
// ---------------------------------------------------------------------------
__global__ __launch_bounds__(256)
void ln_kernel(const float* __restrict__ in, const float* __restrict__ g,
               const float* __restrict__ b, float* __restrict__ xf,
               __bf16* __restrict__ xh) {
  __shared__ float red[D_N];
  int s = blockIdx.x, d = threadIdx.x;
  float v = in[(size_t)s * D_N + d];
  red[d] = v; __syncthreads();
  for (int st = 128; st > 0; st >>= 1) {
    if (d < st) red[d] += red[d + st];
    __syncthreads();
  }
  float mean = red[0] * (1.f / D_N);
  __syncthreads();
  float c = v - mean;
  red[d] = c * c; __syncthreads();
  for (int st = 128; st > 0; st >>= 1) {
    if (d < st) red[d] += red[d + st];
    __syncthreads();
  }
  float var = red[0] * (1.f / D_N);
  float o = c * rsqrtf(var + 1e-5f) * g[d] + b[d];
  xf[(size_t)s * D_N + d] = o;
  xh[(size_t)s * D_N + d] = (__bf16)o;
}

// ---------------------------------------------------------------------------
// Tiny actor/critic heads (fp32 VALU; negligible FLOPs).
// out[0]=value, out[1..16]=policy, out[17..272]=tanh(obs)
// ---------------------------------------------------------------------------
__global__ __launch_bounds__(512)
void heads_kernel(const float* __restrict__ xf, const float* __restrict__ obs,
                  const float* __restrict__ aW1, const float* __restrict__ ab1,
                  const float* __restrict__ aW2, const float* __restrict__ ab2,
                  const float* __restrict__ aWo, const float* __restrict__ abo,
                  const float* __restrict__ cW1, const float* __restrict__ cb1,
                  const float* __restrict__ cW2, const float* __restrict__ cb2,
                  const float* __restrict__ cWo, const float* __restrict__ cbo,
                  float* __restrict__ out) {
  __shared__ float t[D_N];
  __shared__ float h1[ACH_N];
  __shared__ float h2[ACH_N];
  int tid = threadIdx.x;
  if (tid < D_N) t[tid] = xf[tid];
  __syncthreads();
  // actor
  { float a = 0.f;
    for (int j = 0; j < D_N; ++j) a += t[j] * aW1[(size_t)j * ACH_N + tid];
    h1[tid] = tanhf(a + ab1[tid]); }
  __syncthreads();
  { float a = 0.f;
    for (int j = 0; j < ACH_N; ++j) a += h1[j] * aW2[(size_t)j * ACH_N + tid];
    h2[tid] = tanhf(a + ab2[tid]); }
  __syncthreads();
  if (tid < A_N) {
    float a = 0.f;
    for (int j = 0; j < ACH_N; ++j) a += h2[j] * aWo[(size_t)j * A_N + tid];
    out[1 + tid] = a + abo[tid];
  }
  __syncthreads();
  // critic
  { float a = 0.f;
    for (int j = 0; j < D_N; ++j) a += t[j] * cW1[(size_t)j * ACH_N + tid];
    h1[tid] = tanhf(a + cb1[tid]); }
  __syncthreads();
  { float a = 0.f;
    for (int j = 0; j < ACH_N; ++j) a += h1[j] * cW2[(size_t)j * ACH_N + tid];
    h2[tid] = tanhf(a + cb2[tid]); }
  __syncthreads();
  if (tid == 0) {
    float a = 0.f;
    for (int j = 0; j < ACH_N; ++j) a += h2[j] * cWo[j];
    out[0] = a + cbo[0];
  }
  if (tid < OBS_N) out[17 + tid] = tanhf(obs[tid]);
}

// ---------------------------------------------------------------------------
extern "C" void kernel_launch(void* const* d_in, const int* in_sizes, int n_in,
                              void* d_out, int out_size, void* d_ws, size_t ws_size,
                              hipStream_t stream) {
  (void)in_sizes; (void)n_in; (void)out_size; (void)ws_size;
  const float* observation = (const float*)d_in[0];
  const float* old_vectors = (const float*)d_in[1];
  const float* W_core = (const float*)d_in[2];
  const float* b_core = (const float*)d_in[3];
  const float* W_memo = (const float*)d_in[4];
  const float* b_memo = (const float*)d_in[5];
  const float* Wq = (const float*)d_in[6];
  const float* bq = (const float*)d_in[7];
  const float* Wk = (const float*)d_in[8];
  const float* bk = (const float*)d_in[9];
  const float* Wv = (const float*)d_in[10];
  const float* bv = (const float*)d_in[11];
  const float* Wo = (const float*)d_in[12];
  const float* bo = (const float*)d_in[13];
  const float* ln1_g = (const float*)d_in[14];
  const float* ln1_b = (const float*)d_in[15];
  const float* ln2_g = (const float*)d_in[16];
  const float* ln2_b = (const float*)d_in[17];
  const float* W_ff1 = (const float*)d_in[18];
  const float* b_ff1 = (const float*)d_in[19];
  const float* W_ff2 = (const float*)d_in[20];
  const float* b_ff2 = (const float*)d_in[21];
  const float* aW1 = (const float*)d_in[22];
  const float* ab1 = (const float*)d_in[23];
  const float* aW2 = (const float*)d_in[24];
  const float* ab2 = (const float*)d_in[25];
  const float* aWo = (const float*)d_in[26];
  const float* abo = (const float*)d_in[27];
  const float* cW1 = (const float*)d_in[28];
  const float* cb1 = (const float*)d_in[29];
  const float* cW2 = (const float*)d_in[30];
  const float* cb2 = (const float*)d_in[31];
  const float* cWo = (const float*)d_in[32];
  const float* cbo = (const float*)d_in[33];

  // ---- workspace carve-up (aligned to 256B) ----
  char* p = (char*)d_ws;
  auto take = [&](size_t bytes) -> void* {
    void* r = (void*)p;
    p += (bytes + 255) & ~(size_t)255;
    return r;
  };
  float*  x_f   = (float*) take((size_t)S_LEN * D_N * 4);
  float*  tmp_f = (float*) take((size_t)S_LEN * D_N * 4);
  __bf16* x_h   = (__bf16*)take((size_t)S_LEN * D_N * 2);
  __bf16* q_h   = (__bf16*)take((size_t)S_LEN * D_N * 2);
  __bf16* k_h   = (__bf16*)take((size_t)S_LEN * D_N * 2);
  __bf16* vT_h  = (__bf16*)take((size_t)D_N * S_LEN * 2);
  __bf16* ctx_h = (__bf16*)take((size_t)S_LEN * D_N * 2);
  __bf16* ff_h  = (__bf16*)take((size_t)S_LEN * FF_N * 2);
  __bf16* wqT   = (__bf16*)take((size_t)L_N * D_N * D_N * 2);
  __bf16* wkT   = (__bf16*)take((size_t)L_N * D_N * D_N * 2);
  __bf16* wvT   = (__bf16*)take((size_t)L_N * D_N * D_N * 2);
  __bf16* woT   = (__bf16*)take((size_t)L_N * D_N * D_N * 2);
  __bf16* wff1T = (__bf16*)take((size_t)L_N * D_N * FF_N * 2);
  __bf16* wff2T = (__bf16*)take((size_t)L_N * FF_N * D_N * 2);

  // ---- weight cast/transpose (fp32 KxN -> bf16 NxK) ----
  for (int l = 0; l < L_N; ++l) {
    cast_transpose_kernel<<<256, 256, 0, stream>>>(Wq + (size_t)l * D_N * D_N,
                                                   wqT + (size_t)l * D_N * D_N, D_N, D_N);
    cast_transpose_kernel<<<256, 256, 0, stream>>>(Wk + (size_t)l * D_N * D_N,
                                                   wkT + (size_t)l * D_N * D_N, D_N, D_N);
    cast_transpose_kernel<<<256, 256, 0, stream>>>(Wv + (size_t)l * D_N * D_N,
                                                   wvT + (size_t)l * D_N * D_N, D_N, D_N);
    cast_transpose_kernel<<<256, 256, 0, stream>>>(Wo + (size_t)l * D_N * D_N,
                                                   woT + (size_t)l * D_N * D_N, D_N, D_N);
    cast_transpose_kernel<<<512, 256, 0, stream>>>(W_ff1 + (size_t)l * D_N * FF_N,
                                                   wff1T + (size_t)l * D_N * FF_N, D_N, FF_N);
    cast_transpose_kernel<<<512, 256, 0, stream>>>(W_ff2 + (size_t)l * FF_N * D_N,
                                                   wff2T + (size_t)l * FF_N * D_N, FF_N, D_N);
  }

  // ---- embedding ----
  embed_kernel<<<S_LEN, 256, 0, stream>>>(observation, old_vectors, W_core, b_core,
                                          W_memo, b_memo, x_f, x_h);

  auto gemm = [&](const __bf16* A, const __bf16* Bt, const float* bias,
                  const float* resid, float scale, int act,
                  float* Cf, __bf16* Cb, __bf16* CbT, int M, int N, int K) {
    dim3 g(N / 64, M / 64);
    gemm_bf16_wmma<<<g, 128, 0, stream>>>(A, Bt, bias, resid, scale, act,
                                          Cf, Cb, CbT, M, N, K);
  };

  const float qscale = 0.17677669529663687f;  // 1/sqrt(32)

  for (int l = 0; l < L_N; ++l) {
    const __bf16* wq = wqT + (size_t)l * D_N * D_N;
    const __bf16* wk = wkT + (size_t)l * D_N * D_N;
    const __bf16* wv = wvT + (size_t)l * D_N * D_N;
    const __bf16* wo = woT + (size_t)l * D_N * D_N;
    const __bf16* w1 = wff1T + (size_t)l * D_N * FF_N;
    const __bf16* w2 = wff2T + (size_t)l * FF_N * D_N;

    // QKV projections (scale folded into Q; V stored dim-major for attention)
    gemm(x_h, wq, bq + l * D_N, nullptr, qscale, 0, nullptr, q_h, nullptr, S_LEN, D_N, D_N);
    gemm(x_h, wk, bk + l * D_N, nullptr, 1.f,    0, nullptr, k_h, nullptr, S_LEN, D_N, D_N);
    gemm(x_h, wv, bv + l * D_N, nullptr, 1.f,    0, nullptr, nullptr, vT_h, S_LEN, D_N, D_N);

    // attention: grid = (heads, query tiles), one wave each
    attn_wmma<<<dim3(H_N, S_LEN / 16), 32, 0, stream>>>(q_h, k_h, vT_h, ctx_h);

    // output projection + residual, then LN1
    gemm(ctx_h, wo, bo + l * D_N, x_f, 1.f, 0, tmp_f, nullptr, nullptr, S_LEN, D_N, D_N);
    ln_kernel<<<S_LEN, 256, 0, stream>>>(tmp_f, ln1_g + l * D_N, ln1_b + l * D_N, x_f, x_h);

    // FFN: gelu(x@W1+b1) @ W2 + b2 + residual, then LN2
    gemm(x_h, w1, b_ff1 + l * FF_N, nullptr, 1.f, 1, nullptr, ff_h, nullptr, S_LEN, FF_N, D_N);
    gemm(ff_h, w2, b_ff2 + l * D_N, x_f, 1.f, 0, tmp_f, nullptr, nullptr, S_LEN, D_N, FF_N);
    ln_kernel<<<S_LEN, 256, 0, stream>>>(tmp_f, ln2_g + l * D_N, ln2_b + l * D_N, x_f, x_h);
  }

  // ---- heads ----
  heads_kernel<<<1, 512, 0, stream>>>(x_f, observation,
                                      aW1, ab1, aW2, ab2, aWo, abo,
                                      cW1, cb1, cW2, cb2, cWo, cbo,
                                      (float*)d_out);
}
